// Attention_28802050687669
// MI455X (gfx1250) — matrix operand order repaired and checked by reading, not compiled, not run on hardware
//
#include <hip/hip_runtime.h>
#include <cstdint>
#include <cstddef>

// ---------- types ----------
typedef unsigned short u16;
typedef unsigned int   u32;
typedef __attribute__((ext_vector_type(4)))  u32    u32x4;
typedef __attribute__((ext_vector_type(8)))  float  v8f;
typedef __attribute__((ext_vector_type(16))) __bf16 v16bf;

union Frag { u32x4 u[2]; v16bf v; };   // 32B: one WMMA A/B operand (16 bf16)

__device__ __forceinline__ u16 f2bf(float f) {
  u32 x = __builtin_bit_cast(u32, f);
  u32 r = (x + 0x7FFFu + ((x >> 16) & 1u)) >> 16;   // RNE
  return (u16)r;
}

__device__ __forceinline__ v8f wmma_bf16(v16bf a, v16bf b, v8f c) {
  return __builtin_amdgcn_wmma_f32_16x16x32_bf16(false, a, false, b, (short)0, c,
                                                 false, false);
}

__device__ __forceinline__ float half_max16(float v) {
  v = fmaxf(v, __shfl_xor(v, 1, 32));
  v = fmaxf(v, __shfl_xor(v, 2, 32));
  v = fmaxf(v, __shfl_xor(v, 4, 32));
  v = fmaxf(v, __shfl_xor(v, 8, 32));
  return v;
}
__device__ __forceinline__ float half_sum16(float v) {
  v += __shfl_xor(v, 1, 32);
  v += __shfl_xor(v, 2, 32);
  v += __shfl_xor(v, 4, 32);
  v += __shfl_xor(v, 8, 32);
  return v;
}

// ---------- Tensor Data Mover (CDNA5 TDM) ----------
#if __has_builtin(__builtin_amdgcn_tensor_load_to_lds)
#define HAVE_TDM 1
#else
#define HAVE_TDM 0
#endif

// 2D bf16 tile: global[tile_d1][tile_d0] rows (stride stride_d0 elems) -> LDS
// at lds_off, with (pad_amount+1) dwords appended every 2^(pad_interval+1)
// dwords = hardware bank-conflict row padding.
__device__ __forceinline__ void tdm_load_2d_bf16(u32 lds_off, const u16* gaddr,
                                                 u32 tile_d0, u32 tile_d1,
                                                 u32 tensor_d0, u32 tensor_d1,
                                                 u32 stride_d0,
                                                 u32 pad_interval, u32 pad_amount) {
#if HAVE_TDM
  typedef __attribute__((ext_vector_type(4))) unsigned int tu4;
  typedef __attribute__((ext_vector_type(8))) int ti8;
  typedef __attribute__((ext_vector_type(4))) int ti4;
  unsigned long long ga = (unsigned long long)(uintptr_t)gaddr;
  tu4 g0 = { 1u,                                  // count=1
             lds_off,                             // lds_addr
             (u32)ga,                             // global_addr[31:0]
             (((u32)(ga >> 32)) & 0x01FFFFFFu) | (2u << 30) };  // addr[56:32]|type=2
  ti8 g1 = { (int)((1u << 16) | (1u << 20) | (pad_interval << 22) | (pad_amount << 25)),
             (int)((tensor_d0 & 0xFFFFu) << 16),
             (int)((tensor_d0 >> 16) | ((tensor_d1 & 0xFFFFu) << 16)),
             (int)((tensor_d1 >> 16) | (tile_d0 << 16)),
             (int)tile_d1,
             (int)stride_d0,
             0, 0 };
  ti4 g2 = {0, 0, 0, 0};
  ti4 g3 = {0, 0, 0, 0};
#if defined(__clang_major__) && __clang_major__ >= 23
  ti8 g4 = {0, 0, 0, 0, 0, 0, 0, 0};
  __builtin_amdgcn_tensor_load_to_lds(g0, g1, g2, g3, g4, 0);
#else
  __builtin_amdgcn_tensor_load_to_lds(g0, g1, g2, g3, 0);
#endif
#else
  (void)lds_off; (void)gaddr; (void)tile_d0; (void)tile_d1;
  (void)tensor_d0; (void)tensor_d1; (void)stride_d0;
  (void)pad_interval; (void)pad_amount;
#endif
}

__device__ __forceinline__ void tdm_wait0() {
#if __has_builtin(__builtin_amdgcn_s_wait_tensorcnt)
  __builtin_amdgcn_s_wait_tensorcnt((short)0);
#endif
}

// ---------- problem constants ----------
#define BB 4
#define TT 1024
#define DD 4096
#define NH 32
#define KH 8
#define HH 128
#define QG_COLS 8192
#define KV_COLS 1024
#define ATT_SCALE 0.08838834764831845f
#define NEG_BIG  -3.0e38f

// ================= fp32 -> bf16 (row-major, for A-side x) =================
__global__ __launch_bounds__(256) void cvt_f32_bf16(const float* __restrict__ in,
                                                    u16* __restrict__ out,
                                                    long long n) {
  long long i = (long long)blockIdx.x * blockDim.x + threadIdx.x;
  long long stride = (long long)gridDim.x * blockDim.x;
  for (; i < n; i += stride) out[i] = f2bf(in[i]);
}

// ========== fp32 [K][N] -> bf16 [N][K] (weights into WMMA-B layout) ==========
__global__ __launch_bounds__(256) void cvt_transpose(const float* __restrict__ in,
                                                     u16* __restrict__ out,
                                                     int K, int N) {
  __shared__ u16 tile[32][33];
  const int tx = threadIdx.x;   // 0..31
  const int ty = threadIdx.y;   // 0..7
  const int n0 = blockIdx.x * 32;
  const int k0 = blockIdx.y * 32;
#pragma unroll
  for (int i = 0; i < 4; ++i)
    tile[ty * 4 + i][tx] = f2bf(in[(size_t)(k0 + ty * 4 + i) * N + n0 + tx]);
  __syncthreads();
#pragma unroll
  for (int i = 0; i < 4; ++i)
    out[(size_t)(n0 + ty * 4 + i) * K + k0 + tx] = tile[tx][ty * 4 + i];
}

// ========== v [B*T][8*128] -> per-(b,kh) planes vt[plane][h][t] ==========
__global__ __launch_bounds__(256) void transpose_v(const u16* __restrict__ v,
                                                   u16* __restrict__ vt) {
  __shared__ u16 tile[32][33];
  const int tx = threadIdx.x, ty = threadIdx.y;
  const int s0 = blockIdx.x * 32;
  const int h0 = blockIdx.y * 32;
  const int plane = blockIdx.z;          // b*8 + kh
  const int b = plane >> 3, kh = plane & 7;
#pragma unroll
  for (int i = 0; i < 4; ++i)
    tile[ty * 4 + i][tx] =
        v[(size_t)(b * TT + s0 + ty * 4 + i) * KV_COLS + kh * HH + h0 + tx];
  __syncthreads();
#pragma unroll
  for (int i = 0; i < 4; ++i)
    vt[(size_t)plane * (HH * TT) + (size_t)(h0 + ty * 4 + i) * TT + s0 + tx] =
        tile[tx][ty * 4 + i];
}

// ================= bf16 WMMA GEMM: C[M,N] = A[M,K] * Bt[N,K]^T =================
// Block tile 128x128, BK=32, 256 thr (8 waves, 32x64/wave).
// BOTH tiles staged by TDM (wave0: A, wave1: B), hardware-padded rows of 40 u16.
__device__ __forceinline__ void store_c(float* p, float v) { *p = v; }
__device__ __forceinline__ void store_c(u16* p, float v)   { *p = f2bf(v); }

template <typename OutT>
__global__ __launch_bounds__(256) void gemm_bf16(const u16* __restrict__ A,
                                                 const u16* __restrict__ Bt,
                                                 OutT* __restrict__ C,
                                                 int M, int N, int K) {
  __shared__ __align__(16) u16 lds_a[2][128][40];  // [m][k]
  __shared__ __align__(16) u16 lds_b[2][128][40];  // [n][k]

  const int tid  = threadIdx.x;
  const int lane = tid & 31;
  const int wid  = tid >> 5;
  const int half = lane >> 4;
  const int l16  = lane & 15;
  const int wm   = wid & 3;
  const int wn   = wid >> 2;
  const int m0   = blockIdx.y * 128;
  const int n0   = blockIdx.x * 128;

  v8f acc[2][4];
  const v8f vzero = {0.f,0.f,0.f,0.f,0.f,0.f,0.f,0.f};
#pragma unroll
  for (int i = 0; i < 2; ++i)
#pragma unroll
    for (int j = 0; j < 4; ++j) acc[i][j] = vzero;

  const int NT = K >> 5;
  const u32 ldsa_base = (u32)(uintptr_t)&lds_a[0][0][0];
  const u32 ldsb_base = (u32)(uintptr_t)&lds_b[0][0][0];
  const u32 LDS_BUF   = 128u * 40u * 2u;

  auto tdm_tiles = [&](int kt, int buf) {
#if HAVE_TDM
    if (wid == 0)
      tdm_load_2d_bf16(ldsa_base + (u32)buf * LDS_BUF,
                       A + (size_t)m0 * K + kt * 32,
                       32, 128, (u32)K, (u32)M, (u32)K, /*pad*/ 3, 3);
    if (wid == 1)
      tdm_load_2d_bf16(ldsb_base + (u32)buf * LDS_BUF,
                       Bt + (size_t)n0 * K + kt * 32,
                       32, 128, (u32)K, (u32)N, (u32)K, /*pad*/ 3, 3);
#else
    // manual fallback: both layouts already fragment-friendly, plain copies
    const int row = tid >> 1, col = (tid & 1) * 16;
    const u16* ap = A  + (size_t)(m0 + row) * K + kt * 32 + col;
    const u16* bp = Bt + (size_t)(n0 + row) * K + kt * 32 + col;
#pragma unroll
    for (int i = 0; i < 2; ++i) {
      *(u32x4*)&lds_a[buf][row][col + i * 8] = *(const u32x4*)(ap + i * 8);
      *(u32x4*)&lds_b[buf][row][col + i * 8] = *(const u32x4*)(bp + i * 8);
    }
#endif
  };
  auto tiles_wait = [&]() {
#if HAVE_TDM
    if (wid < 2) tdm_wait0();
#endif
  };
  auto compute = [&](int buf) {
    Frag af[2], bf[4];
#pragma unroll
    for (int fm = 0; fm < 2; ++fm) {
      int r = wm * 32 + fm * 16 + l16;
      af[fm].u[0] = *(const u32x4*)&lds_a[buf][r][half * 8];
      af[fm].u[1] = *(const u32x4*)&lds_a[buf][r][16 + half * 8];
    }
#pragma unroll
    for (int fn = 0; fn < 4; ++fn) {
      int c = wn * 64 + fn * 16 + l16;
      bf[fn].u[0] = *(const u32x4*)&lds_b[buf][c][half * 8];
      bf[fn].u[1] = *(const u32x4*)&lds_b[buf][c][16 + half * 8];
    }
#pragma unroll
    for (int fm = 0; fm < 2; ++fm)
#pragma unroll
      for (int fn = 0; fn < 4; ++fn)
        acc[fm][fn] = wmma_bf16(af[fm].v, bf[fn].v, acc[fm][fn]);
  };

  tdm_tiles(0, 0);
  tiles_wait();
  __syncthreads();

  for (int kt = 0; kt < NT; ++kt) {
    const int buf = kt & 1;
    if (kt + 1 < NT) {
      tdm_tiles(kt + 1, buf ^ 1);   // DMA next tiles, overlapped with WMMAs
      if (kt + 2 < NT) {            // speculative L2 warm -> global_prefetch_b8
        __builtin_prefetch(A  + (size_t)(m0 + (tid >> 1)) * K + (kt + 2) * 32, 0, 1);
        __builtin_prefetch(Bt + (size_t)(n0 + (tid >> 1)) * K + (kt + 2) * 32, 0, 1);
      }
    }
    compute(buf);
    if (kt + 1 < NT) {
      tiles_wait();                 // publish DMA'd tiles before the barrier
      __syncthreads();
    }
  }

#pragma unroll
  for (int fm = 0; fm < 2; ++fm)
#pragma unroll
    for (int fn = 0; fn < 4; ++fn)
#pragma unroll
      for (int j = 0; j < 8; ++j) {
        int gr = m0 + wm * 32 + fm * 16 + j + 8 * half;
        int gc = n0 + wn * 64 + fn * 16 + l16;
        store_c(&C[(size_t)gr * N + gc], acc[fm][fn][j]);
      }
}

// ================= fused RMS-norm + partial RoPE (80 dims) -> bf16 =================
__global__ __launch_bounds__(128) void normrope_kernel(const float* __restrict__ in,
                                                       const float* __restrict__ w,
                                                       const int* __restrict__ pos,
                                                       u16* __restrict__ out,
                                                       int in_row_stride,
                                                       int in_head_stride,
                                                       int nheads) {
  const int head = blockIdx.x;
  const int row  = blockIdx.y;
  const int h    = threadIdx.x;
  __shared__ float red[128];
  __shared__ float ybuf[128];

  float x = in[(size_t)row * in_row_stride + head * in_head_stride + h];
  red[h] = x * x;
  __syncthreads();
#pragma unroll
  for (int off = 64; off > 0; off >>= 1) {
    if (h < off) red[h] += red[h + off];
    __syncthreads();
  }
  float r = rsqrtf(red[0] * (1.0f / 128.0f) + 1e-6f);
  float y = x * r * (1.0f + w[h]);
  ybuf[h] = y;
  __syncthreads();

  float val;
  if (h < 80) {
    const float p = (float)pos[row];
    const int i = (h < 40) ? h : (h - 40);
    const float ts = __powf(1.0e6f, (2.0f * (float)i) / 80.0f);
    const float a  = p / ts;
    const float sn = sinf(a), cs = cosf(a);
    val = (h < 40) ? (y * cs - ybuf[h + 40] * sn)
                   : (y * cs + ybuf[h - 40] * sn);
  } else {
    val = y;
  }
  out[(size_t)row * (nheads * HH) + head * HH + h] = f2bf(val);
}

// ================= flash attention: K and V tiles both via TDM ================
// grid: (T/128, 32 heads, B). block: 256 (8 waves x 16 q-rows). S-tile = 64.
__global__ __launch_bounds__(256) void flash_attn_kernel(const u16* __restrict__ qb,
                                                         const u16* __restrict__ kb,
                                                         const u16* __restrict__ vt,
                                                         const float* __restrict__ qg,
                                                         u16* __restrict__ og) {
  __shared__ __align__(16) u16 lds_k [64][136];     // [s][h]
  __shared__ __align__(16) u16 lds_vt[128][72];     // [h][s]
  __shared__ __align__(16) u16 lds_p [8][16][40];   // per-wave P re-striping

  const int tid  = threadIdx.x;
  const int lane = tid & 31;
  const int wid  = tid >> 5;
  const int half = lane >> 4;
  const int l16  = lane & 15;

  const int b  = blockIdx.z;
  const int n  = blockIdx.y;
  const int kh = n >> 2;
  const int t0 = blockIdx.x * 128;
  const int rwave = t0 + wid * 16;

  const u32 ldsk_base  = (u32)(uintptr_t)&lds_k[0][0];
  const u32 ldsvt_base = (u32)(uintptr_t)&lds_vt[0][0];
  const u16* vplane = vt + (size_t)(b * KH + kh) * (HH * TT);

  Frag qfrag[4];
  {
    const int qr = rwave + l16;
    const u16* qp = qb + (size_t)(b * TT + qr) * DD + n * HH;
#pragma unroll
    for (int hs = 0; hs < 4; ++hs) {
      qfrag[hs].u[0] = *(const u32x4*)(qp + hs * 32 + half * 8);
      qfrag[hs].u[1] = *(const u32x4*)(qp + hs * 32 + 16 + half * 8);
    }
  }

  const v8f vzero = {0.f,0.f,0.f,0.f,0.f,0.f,0.f,0.f};
  v8f acc_o[8];
#pragma unroll
  for (int i = 0; i < 8; ++i) acc_o[i] = vzero;
  float m_j[8], l_j[8];
#pragma unroll
  for (int j = 0; j < 8; ++j) { m_j[j] = NEG_BIG; l_j[j] = 0.f; }

  const int nt = (t0 >> 6) + 2;
  for (int st = 0; st < nt; ++st) {
    const int s0 = st * 64;
    __syncthreads();  // previous tile fully consumed

#if HAVE_TDM
    if (wid == 0)  // K tile 64x128, rows padded 256B+16B -> [64][136]
      tdm_load_2d_bf16(ldsk_base,
                       kb + (size_t)(b * TT + s0) * KV_COLS + kh * HH,
                       HH, 64, KV_COLS, BB * TT, KV_COLS, /*pad*/ 5, 3);
    if (wid == 1)  // V^T tile 128x64, rows padded 128B+16B -> [128][72]
      tdm_load_2d_bf16(ldsvt_base,
                       vplane + s0,
                       64, HH, TT, BB * KH * HH, TT, /*pad*/ 4, 3);
    if (wid < 2) tdm_wait0();
#else
    {
      const int sl = tid >> 2, part = tid & 3;
      const size_t gk = (size_t)(b * TT + s0 + sl) * KV_COLS + kh * HH + part * 32;
#pragma unroll
      for (int i = 0; i < 4; ++i)
        *(u32x4*)&lds_k[sl][part * 32 + i * 8] = *(const u32x4*)(kb + gk + i * 8);
      const int hr = tid >> 1, sc = (tid & 1) * 32;
      const u16* vp = vplane + (size_t)hr * TT + s0 + sc;
#pragma unroll
      for (int i = 0; i < 4; ++i)
        *(u32x4*)&lds_vt[hr][sc + i * 8] = *(const u32x4*)(vp + i * 8);
    }
#endif
    __syncthreads();

    // ---- S = Q * K^T ----
    v8f s_acc[4];
#pragma unroll
    for (int sf = 0; sf < 4; ++sf) s_acc[sf] = vzero;
#pragma unroll
    for (int hs = 0; hs < 4; ++hs) {
#pragma unroll
      for (int sf = 0; sf < 4; ++sf) {
        Frag kf;
        kf.u[0] = *(const u32x4*)&lds_k[sf * 16 + l16][hs * 32 + half * 8];
        kf.u[1] = *(const u32x4*)&lds_k[sf * 16 + l16][hs * 32 + 16 + half * 8];
        s_acc[sf] = wmma_bf16(qfrag[hs].v, kf.v, s_acc[sf]);
      }
    }

    // ---- causal mask + scale ----
#pragma unroll
    for (int sf = 0; sf < 4; ++sf) {
      const int sc = s0 + sf * 16 + l16;
#pragma unroll
      for (int j = 0; j < 8; ++j) {
        const int tr = rwave + j + 8 * half;
        float v = s_acc[sf][j] * ATT_SCALE;
        if (sc > tr) v = NEG_BIG;
        s_acc[sf][j] = v;
      }
    }

    // ---- online softmax ----
#pragma unroll
    for (int j = 0; j < 8; ++j) {
      float rmax = fmaxf(fmaxf(s_acc[0][j], s_acc[1][j]),
                         fmaxf(s_acc[2][j], s_acc[3][j]));
      rmax = half_max16(rmax);
      const float mnew = fmaxf(m_j[j], rmax);
      const float corr = __expf(m_j[j] - mnew);
      float rsum = 0.f;
#pragma unroll
      for (int sf = 0; sf < 4; ++sf) {
        float p = __expf(s_acc[sf][j] - mnew);
        s_acc[sf][j] = p;
        rsum += p;
      }
      rsum = half_sum16(rsum);
      l_j[j] = l_j[j] * corr + rsum;
      m_j[j] = mnew;
#pragma unroll
      for (int hf = 0; hf < 8; ++hf) acc_o[hf][j] *= corr;
    }

    // ---- O += P * V ----
#pragma unroll
    for (int ks = 0; ks < 2; ++ks) {
#pragma unroll
      for (int c = 0; c < 2; ++c) {
        const int sf = ks * 2 + c;
#pragma unroll
        for (int j = 0; j < 8; ++j)
          lds_p[wid][j + 8 * half][c * 16 + l16] = f2bf(s_acc[sf][j]);
      }
      Frag pf;
      pf.u[0] = *(const u32x4*)&lds_p[wid][l16][half * 8];
      pf.u[1] = *(const u32x4*)&lds_p[wid][l16][16 + half * 8];
#pragma unroll
      for (int hf = 0; hf < 8; ++hf) {
        Frag vf;
        vf.u[0] = *(const u32x4*)&lds_vt[hf * 16 + l16][ks * 32 + half * 8];
        vf.u[1] = *(const u32x4*)&lds_vt[hf * 16 + l16][ks * 32 + 16 + half * 8];
        acc_o[hf] = wmma_bf16(pf.v, vf.v, acc_o[hf]);
      }
    }
  }

  // ---- epilogue ----
#pragma unroll
  for (int hf = 0; hf < 8; ++hf)
#pragma unroll
    for (int j = 0; j < 8; ++j) {
      const int tr = rwave + j + 8 * half;
      const int hc = hf * 16 + l16;
      float o = acc_o[hf][j] / l_j[j];
      const float g = qg[(size_t)(b * TT + tr) * QG_COLS + n * 256 + 128 + hc];
      o *= 1.0f / (1.0f + __expf(-g));
      og[(size_t)(b * TT + tr) * DD + n * HH + hc] = f2bf(o);
    }
}

// ================= orchestration =================
extern "C" void kernel_launch(void* const* d_in, const int* in_sizes, int n_in,
                              void* d_out, int out_size, void* d_ws, size_t ws_size,
                              hipStream_t stream) {
  const float* x   = (const float*)d_in[0];
  const int*   pos = (const int*)d_in[1];
  const float* wq  = (const float*)d_in[2];
  const float* wk  = (const float*)d_in[3];
  const float* wv  = (const float*)d_in[4];
  const float* wo  = (const float*)d_in[5];
  const float* qnw = (const float*)d_in[6];
  const float* knw = (const float*)d_in[7];
  float* out = (float*)d_out;

  const long long NXB = (long long)BB * TT * DD;
  const long long NWQ = (long long)DD * QG_COLS;
  const long long NWK = (long long)DD * KV_COLS;
  const long long NWO = (long long)DD * DD;
  const long long NQG = (long long)BB * TT * QG_COLS;
  const long long NKV = (long long)BB * TT * KV_COLS;

  char* ws = (char*)d_ws;
  size_t off = 0;
  auto alloc = [&](size_t bytes) { void* p = ws + off; off += (bytes + 255) & ~(size_t)255; return p; };
  u16*   xb  = (u16*)alloc(NXB * 2);    // x, bf16 row-major (A operand)
  u16*   wqt = (u16*)alloc(NWQ * 2);    // wq^T  [8192][4096] bf16
  u16*   wkt = (u16*)alloc(NWK * 2);    // wk^T  [1024][4096]
  u16*   wvt = (u16*)alloc(NWK * 2);    // wv^T  [1024][4096]
  u16*   wot = (u16*)alloc(NWO * 2);    // wo^T  [4096][4096]
  float* qgf = (float*)alloc(NQG * 4);  // q|gate fp32
  float* kf  = (float*)alloc(NKV * 4);  // k fp32
  u16*   vbf = (u16*)alloc(NKV * 2);    // v bf16 row-major
  u16*   vtp = (u16*)alloc(NKV * 2);    // v^T planes [b*8+kh][128][1024]
  u16*   qbb = (u16*)alloc(NXB * 2);    // q bf16 post norm+rope
  u16*   kbb = (u16*)alloc(NKV * 2);    // k bf16 post norm+rope
  u16*   ogb = (u16*)alloc(NXB * 2);    // gated attention out bf16
  (void)ws_size; (void)in_sizes; (void)n_in; (void)out_size;

  // 1) conversions: x row-major; weights converted + transposed into B layout
  cvt_f32_bf16<<<4096, 256, 0, stream>>>(x, xb, NXB);
  cvt_transpose<<<dim3(QG_COLS / 32, DD / 32), dim3(32, 8), 0, stream>>>(wq, wqt, DD, QG_COLS);
  cvt_transpose<<<dim3(KV_COLS / 32, DD / 32), dim3(32, 8), 0, stream>>>(wk, wkt, DD, KV_COLS);
  cvt_transpose<<<dim3(KV_COLS / 32, DD / 32), dim3(32, 8), 0, stream>>>(wv, wvt, DD, KV_COLS);
  cvt_transpose<<<dim3(DD / 32, DD / 32), dim3(32, 8), 0, stream>>>(wo, wot, DD, DD);

  // 2) projections (bf16 WMMA, f32 accumulate, TDM-staged tiles)
  gemm_bf16<float><<<dim3(QG_COLS / 128, (BB * TT) / 128), 256, 0, stream>>>(
      xb, wqt, qgf, BB * TT, QG_COLS, DD);
  gemm_bf16<float><<<dim3(KV_COLS / 128, (BB * TT) / 128), 256, 0, stream>>>(
      xb, wkt, kf, BB * TT, KV_COLS, DD);
  gemm_bf16<u16><<<dim3(KV_COLS / 128, (BB * TT) / 128), 256, 0, stream>>>(
      xb, wvt, vbf, BB * TT, KV_COLS, DD);

  // 3) RMS-norm + RoPE(80); V transposed once into per-head planes
  normrope_kernel<<<dim3(NH, BB * TT), 128, 0, stream>>>(
      qgf, qnw, pos, qbb, QG_COLS, 256, NH);
  normrope_kernel<<<dim3(KH, BB * TT), 128, 0, stream>>>(
      kf, knw, pos, kbb, KV_COLS, 128, KH);
  transpose_v<<<dim3(TT / 32, HH / 32, BB * KH), dim3(32, 8), 0, stream>>>(vbf, vtp);

  // 4) flash attention + sigmoid gate
  flash_attn_kernel<<<dim3(TT / 128, NH, BB), 256, 0, stream>>>(
      qbb, kbb, vtp, qgf, ogb);

  // 5) output projection -> fp32 d_out
  gemm_bf16<float><<<dim3(DD / 128, (BB * TT) / 128), 256, 0, stream>>>(
      ogb, wot, out, BB * TT, DD, DD);
}
